// SMI_62921270886476
// MI455X (gfx1250) — matrix operand order repaired and checked by reading, not compiled, not run on hardware
//
#include <hip/hip_runtime.h>
#include <math.h>

// ---------------------------------------------------------------------------
// SMI pipeline for gfx1250 (MI455X).
// All dense projections go through v_wmma_f32_16x16x32_f16 (wave32 WMMA,
// fp32 accumulate). Attention / softmax / gating epilogues are fused
// VALU+LDS kernels. Shapes: B=16 L=32 K=4 LQ=20 D=512 DL=256 H=8.
// ---------------------------------------------------------------------------

typedef __attribute__((ext_vector_type(16))) _Float16 v16h;
typedef __attribute__((ext_vector_type(8)))  float    v8f;

// ============================ WMMA GEMM ====================================
// C[M,N] = A[M,K] @ W[K,N] + bias[N]   (row-major, fp32 in/out, f16 WMMA)
// block = 256 threads = 8 waves; wave w computes a 16x32 strip (two 16x16
// tiles sharing one A fragment) at m0 = 16*blockIdx.y,
// n0 = 32*(8*blockIdx.x + w). N % 256 == 0, K % 32 == 0, M % 16 == 0 here.
// All offsets are 32-bit (largest buffer is 33.5M elements) so the backend
// can use scalar-base + voffset global addressing.
template <bool ACC>
__global__ __launch_bounds__(256)
void k_gemm(const float* __restrict__ A, const float* __restrict__ W,
            const float* __restrict__ bias, float* __restrict__ C,
            int M, int N, int K) {
  const unsigned wave = threadIdx.x >> 5;
  const unsigned lane = threadIdx.x & 31;
  const unsigned n0 = (blockIdx.x * 8 + wave) * 32;
  const unsigned m0 = blockIdx.y * 16;
  if (n0 >= (unsigned)N) return;             // wave-uniform; EXEC stays all-1s
  const unsigned mrow = m0 + (lane & 15);
  const unsigned ncol = n0 + (lane & 15);
  const unsigned ka = (lane < 16) ? 0u : 16u; // A frag: contiguous k = 16h + s
  const unsigned kb = (lane < 16) ? 0u : 16u; // B frag: contiguous k = 16h + s, same as A
  const unsigned uN = (unsigned)N;
  unsigned aoff = mrow * (unsigned)K + ka;    // element offsets, 32-bit
  unsigned woff = kb * uN + ncol;
  v8f acc0 = {}, acc1 = {};
  for (int k0 = 0; k0 < K; k0 += 32) {
    v16h a, b0, b1;
#pragma unroll
    for (int j = 0; j < 8; ++j) a[j]     = (_Float16)A[aoff + j];      // K=ka+j
#pragma unroll
    for (int j = 0; j < 8; ++j) a[8 + j] = (_Float16)A[aoff + 8 + j];  // K=ka+8+j
#pragma unroll
    for (int j = 0; j < 16; ++j) {
      b0[j] = (_Float16)W[woff + (unsigned)j * uN];                    // K=kb+j
      b1[j] = (_Float16)W[woff + (unsigned)j * uN + 16];
    }
    if (k0 + 32 < K) {                       // hint next tiles into cache
      __builtin_prefetch(A + aoff + 32, 0, 1);
      __builtin_prefetch(W + woff + 32u * uN, 0, 1);
    }
    acc0 = __builtin_amdgcn_wmma_f32_16x16x32_f16(
        false, a, false, b0, (short)0, acc0, false, false);
    acc1 = __builtin_amdgcn_wmma_f32_16x16x32_f16(
        false, a, false, b1, (short)0, acc1, false, false);
    aoff += 32u;
    woff += 32u * uN;
  }
  const float bv0 = bias[ncol];
  const float bv1 = bias[ncol + 16];
  const unsigned mb = m0 + ((lane < 16) ? 0u : 8u); // C frag: reg r -> M = mb+r
#pragma unroll
  for (int r = 0; r < 8; ++r) {
    unsigned i0 = (mb + (unsigned)r) * uN + ncol;
    float v0 = acc0[r] + bv0;
    float v1 = acc1[r] + bv1;
    if (ACC) {
      C[i0]      += v0;
      C[i0 + 16] += v1;
    } else {
      C[i0]      = v0;
      C[i0 + 16] = v1;
    }
  }
}

// ===================== generic multi-head attention ========================
// grid = (nq, H=8, B=16). Q/K rows: (b*nq+qi)*512 (+64*h), V same layout.
// kmask[b*nk+k] int; qmask (optional): out *= m^2 (cga query-mask rule).
__global__ __launch_bounds__(256)
void k_attn(const float* __restrict__ Q, const float* __restrict__ Kp,
            const float* __restrict__ V, float* __restrict__ O,
            const int* __restrict__ kmask, const int* __restrict__ qmask,
            int nq, int nk, float scale) {
  const int qi = blockIdx.x, h = blockIdx.y, b = blockIdx.z;
  const int tid = threadIdx.x;
  __shared__ float lg[1024];
  __shared__ float qs[64];
  __shared__ float rb[256];
  __shared__ float s_mx, s_sum;
  const unsigned qoff = (unsigned)(b * nq + qi) * 512u + h * 64u;
  if (tid < 64) qs[tid] = Q[qoff + tid];
  __syncthreads();
  float pmax = -3.0e38f;
  for (int k = tid; k < nk; k += 256) {
    const float* kr = Kp + (unsigned)(b * nk + k) * 512u + h * 64u;
    float s = 0.f;
    for (int d = 0; d < 64; ++d) s += qs[d] * kr[d];
    s *= scale;
    int m = kmask[b * nk + k];
    s = (m == 0) ? -1.0e9f : s * (float)m;
    lg[k] = s;
    pmax = fmaxf(pmax, s);
  }
  rb[tid] = pmax; __syncthreads();
  for (int s = 128; s > 0; s >>= 1) {
    if (tid < s) rb[tid] = fmaxf(rb[tid], rb[tid + s]);
    __syncthreads();
  }
  if (tid == 0) s_mx = rb[0];
  __syncthreads();
  float psum = 0.f;
  for (int k = tid; k < nk; k += 256) {
    float e = expf(lg[k] - s_mx);
    lg[k] = e; psum += e;
  }
  rb[tid] = psum; __syncthreads();
  for (int s = 128; s > 0; s >>= 1) {
    if (tid < s) rb[tid] += rb[tid + s];
    __syncthreads();
  }
  if (tid == 0) s_sum = rb[0];
  __syncthreads();
  // output: 4 key-chunks x 64 dims, then LDS reduce over chunks
  {
    const int d = tid & 63;
    const int c = tid >> 6;
    float po = 0.f;
    for (int k = c; k < nk; k += 4)
      po += lg[k] * V[(unsigned)(b * nk + k) * 512u + h * 64u + d];
    rb[tid] = po;
  }
  __syncthreads();
  if (tid < 64) {
    float o = (rb[tid] + rb[tid + 64] + rb[tid + 128] + rb[tid + 192]) / s_sum;
    if (qmask) { float mq = (float)qmask[b * nq + qi]; o *= mq * mq; }
    O[qoff + tid] = o;
  }
}

// ======================= small element-wise kernels ========================
__global__ void k_mask_rows(float* __restrict__ X, const int* __restrict__ mask,
                            int shift, int cols, size_t total) {
  size_t i = (size_t)blockIdx.x * 256 + threadIdx.x;
  if (i >= total) return;
  size_t r = i / (size_t)cols;
  X[i] *= (float)mask[r >> shift];
}

// f_caq row: 20-way softmax over word keys + weighted sum of f_w_hat, *fcm.
__global__ __launch_bounds__(64)
void k_cu_caq(const float* __restrict__ qh, const float* __restrict__ kh,
              const float* __restrict__ fwh, const int* __restrict__ qm,
              const int* __restrict__ mm, float* __restrict__ caq) {
  const int r = blockIdx.x;          // b*4096 + x*128 + y*4 + k
  const int b = r >> 12;
  const int tid = threadIdx.x;
  __shared__ float q[256];
  __shared__ float lgv[20];
  __shared__ float p[20];
  for (int i = tid; i < 256; i += 64) q[i] = qh[(unsigned)r * 256u + i];
  __syncthreads();
  if (tid < 20) {
    const float* kr = kh + (unsigned)(b * 20 + tid) * 256u;
    float s = 0.f;
    for (int d = 0; d < 256; ++d) s += q[d] * kr[d];
    s *= 0.0625f;                                   // 1/sqrt(256)
    int m = qm[b * 20 + tid];
    lgv[tid] = (m == 0) ? -1.0e9f : s * (float)m;
  }
  __syncthreads();
  if (tid == 0) {
    float mx = lgv[0];
    for (int w = 1; w < 20; ++w) mx = fmaxf(mx, lgv[w]);
    float sum = 0.f;
    for (int w = 0; w < 20; ++w) { float e = expf(lgv[w] - mx); p[w] = e; sum += e; }
    float inv = 1.0f / sum;
    for (int w = 0; w < 20; ++w) p[w] *= inv;
  }
  __syncthreads();
  const float mf = (float)mm[r >> 2];               // fcm (moment mask)
  for (int d = tid; d < 256; d += 64) {
    float s = 0.f;
    for (int w = 0; w < 20; ++w) s += p[w] * fwh[(unsigned)(b * 20 + w) * 256u + d];
    caq[(unsigned)r * 256u + d] = s * mf;
  }
}

// f_cq = f_c_hat * (f_caq + f_s_hat broadcast)
__global__ void k_cu_fcq(const float* __restrict__ fchat, const float* __restrict__ caq,
                         const float* __restrict__ shat, float* __restrict__ out,
                         size_t total) {
  size_t i = (size_t)blockIdx.x * 256 + threadIdx.x;
  if (i >= total) return;
  size_t r = i >> 8;
  size_t b = r >> 12;
  int d = (int)(i & 255);
  out[i] = fchat[i] * (caq[i] + shat[b * 256 + d]);
}

// a_c = softmax_j(f_cq f_cq^T / 16) * fcm ; f_cc_pre = a_c @ f_c_hat  (4x4 tiles)
__global__ __launch_bounds__(128)
void k_cu_ac(const float* __restrict__ fcq, const float* __restrict__ fchat,
             const int* __restrict__ mm, float* __restrict__ fccpre) {
  const int t = blockIdx.x;                         // b*1024 + xy
  const int tid = threadIdx.x;
  __shared__ float q[4][256];
  __shared__ float ch[4][256];
  __shared__ float a[4][4];
  const unsigned base = (unsigned)t * 1024u;
  for (int i = tid; i < 1024; i += 128) {
    q[i >> 8][i & 255]  = fcq[base + i];
    ch[i >> 8][i & 255] = fchat[base + i];
  }
  __syncthreads();
  if (tid < 16) {
    int i = tid >> 2, j = tid & 3;
    float s = 0.f;
    for (int d = 0; d < 256; ++d) s += q[i][d] * q[j][d];
    a[i][j] = s * 0.0625f;
  }
  __syncthreads();
  if (tid < 4) {
    const float mf = (float)mm[t];
    int i = tid;
    float mx = a[i][0];
    for (int j = 1; j < 4; ++j) mx = fmaxf(mx, a[i][j]);
    float e[4], sum = 0.f;
    for (int j = 0; j < 4; ++j) { e[j] = expf(a[i][j] - mx); sum += e[j]; }
    for (int j = 0; j < 4; ++j) a[i][j] = e[j] / sum * mf;   // softmax then *fcm
  }
  __syncthreads();
  for (int o = tid; o < 1024; o += 128) {
    int i = o >> 8, d = o & 255;
    float s = 0.f;
    for (int j = 0; j < 4; ++j) s += a[i][j] * ch[j][d];
    fccpre[base + o] = s;
  }
}

// fc_out = f_cc*fcm + fc + sigmoid(fm*fs)*fm   (in-place on c-projection)
__global__ void k_cu_epi(float* __restrict__ tgt, const float* __restrict__ fc,
                         const float* __restrict__ fm, const float* __restrict__ fsu,
                         const int* __restrict__ mm, size_t total) {
  size_t i = (size_t)blockIdx.x * 256 + threadIdx.x;
  if (i >= total) return;
  const size_t d  = i & 511;
  const size_t rk = i >> 9;                        // b*4096 + idx
  const size_t bxy = rk >> 2;                      // b*1024 + xy
  const float mf  = (float)mm[bxy];
  const float fmv = fm[bxy * 512 + d];
  const float fsv = fsu[(rk >> 12) * 512 + d];
  const float g = 1.f / (1.f + expf(-fmv * fsv));
  tgt[i] = tgt[i] * mf + fc[i] + g * fmv;
}

// f_bq = fb * (f_baq*lm + fs broadcast)
__global__ void k_bu_fbq(const float* __restrict__ fb, const float* __restrict__ baq,
                         const float* __restrict__ fsu, const int* __restrict__ lm,
                         float* __restrict__ bq, size_t total) {
  size_t i = (size_t)blockIdx.x * 256 + threadIdx.x;
  if (i >= total) return;
  size_t bi = i >> 9;
  size_t b = bi >> 5;
  size_t d = i & 511;
  bq[i] = fb[i] * (baq[i] * (float)lm[bi] + fsu[b * 512 + d]);
}

// boundary self-attn row: a_b softmax + f_bb + f_b + f_bm (with sigmoid gate)
__global__ __launch_bounds__(256)
void k_bu_out(const float* __restrict__ fbq, const float* __restrict__ fb,
              const float* __restrict__ fm, const float* __restrict__ fsu,
              const int* __restrict__ lm, float* __restrict__ out) {
  const int bi = blockIdx.x;                       // b*32 + i
  const int b = bi >> 5;
  const int tid = threadIdx.x;
  __shared__ float qrow[512];
  __shared__ float lgv[32];
  __shared__ float p[32];
  for (int d = tid; d < 512; d += 256) qrow[d] = fbq[(unsigned)bi * 512u + d];
  __syncthreads();
  if (tid < 32) {
    const float* kr = fbq + (unsigned)(b * 32 + tid) * 512u;
    float s = 0.f;
    for (int d = 0; d < 512; ++d) s += qrow[d] * kr[d];
    s *= 0.044194173824159216f;                    // 1/sqrt(512)
    int m = lm[b * 32 + tid];
    lgv[tid] = (m == 0) ? -1.0e9f : s * (float)m;
  }
  __syncthreads();
  const float lmI = (float)lm[bi];
  if (tid == 0) {
    float mx = lgv[0];
    for (int j = 1; j < 32; ++j) mx = fmaxf(mx, lgv[j]);
    float sum = 0.f;
    for (int j = 0; j < 32; ++j) { float e = expf(lgv[j] - mx); p[j] = e; sum += e; }
    float inv = lmI / sum;                         // a_b row already * fbm(i)
    for (int j = 0; j < 32; ++j) p[j] *= inv;
  }
  __syncthreads();
  for (int d = tid; d < 512; d += 256) {
    const float fsv = fsu[b * 512 + d];
    float s_bb = 0.f, s_bm = 0.f;
    for (int j = 0; j < 32; ++j) {
      float pj = p[j];
      s_bb += pj * fb[(unsigned)(b * 32 + j) * 512u + d];
      float fmv = fm[((unsigned)bi * 32u + j) * 512u + d];
      float g = 1.f / (1.f + expf(-fmv * fsv));
      s_bm += pj * g * fmv;
    }
    out[(unsigned)bi * 512u + d] = lmI * s_bb + fb[(unsigned)bi * 512u + d] + s_bm;
  }
}

// fb2[b,i,j,d] = fb[b,i,d]*fb[b,j,d]
__global__ void k_outer(const float* __restrict__ fbv, float* __restrict__ o, size_t total) {
  size_t i = (size_t)blockIdx.x * 256 + threadIdx.x;
  if (i >= total) return;
  size_t d = i & 511;
  size_t t = i >> 9;
  size_t b = t >> 10;
  size_t ij = t & 1023;
  size_t ii = ij >> 5, jj = ij & 31;
  o[i] = fbv[(b * 32 + ii) * 512 + d] * fbv[(b * 32 + jj) * 512 + d];
}

// mean over K=4
__global__ void k_meank(const float* __restrict__ fcv, float* __restrict__ o, size_t total) {
  size_t i = (size_t)blockIdx.x * 256 + threadIdx.x;
  if (i >= total) return;
  size_t d = i & 511;
  size_t t = i >> 9;
  const float* p = fcv + (size_t)t * 4 * 512 + d;
  o[i] = 0.25f * (p[0] + p[512] + p[1024] + p[1536]);
}

// fm_out = mm*(conv_fb+conv_fc) + fm   (conv holds accumulated GEMMs)
__global__ void k_mu_epi(const float* __restrict__ conv, const float* __restrict__ fm,
                         const int* __restrict__ mm, float* __restrict__ tgt, size_t total) {
  size_t i = (size_t)blockIdx.x * 256 + threadIdx.x;
  if (i >= total) return;
  tgt[i] = (float)mm[i >> 9] * conv[i] + fm[i];
}

__global__ void k_max(float* __restrict__ dst, const float* __restrict__ src, size_t total) {
  size_t i = (size_t)blockIdx.x * 256 + threadIdx.x;
  if (i >= total) return;
  dst[i] = fmaxf(dst[i], src[i]);
}

// =========================== host helpers ==================================
static inline void gemm(hipStream_t st, const float* A, const float* W,
                        const float* b, float* C, int M, int N, int K, int acc = 0) {
  dim3 g(N / 256, M / 16);
  if (g.x < 1) g.x = 1;
  if (acc)
    k_gemm<true><<<g, 256, 0, st>>>(A, W, b, C, M, N, K);
  else
    k_gemm<false><<<g, 256, 0, st>>>(A, W, b, C, M, N, K);
}

// content unit: p = {akW,akb,aqW,aqb,cW,cb,c_hatW,c_hatb,s_hatW,s_hatb,w_hatW,w_hatb}
static void run_cu(hipStream_t st, const float* fc, const float* fm,
                   const float* fwu, const float* fsu,
                   const int* qm, const int* mm, const float* const* p,
                   float* wsA, float* wsB, float* wsC,
                   float* wh, float* kh, float* sh, float* tgt) {
  gemm(st, fc, p[6], p[7], wsA, 65536, 256, 512);                       // f_c_hat
  k_mask_rows<<<65536, 256, 0, st>>>(wsA, mm, 2, 256, (size_t)65536 * 256);
  gemm(st, fwu, p[10], p[11], wh, 320, 256, 512);                       // f_w_hat
  k_mask_rows<<<320, 256, 0, st>>>(wh, qm, 0, 256, (size_t)320 * 256);
  gemm(st, fsu, p[8], p[9], sh, 16, 256, 512);                          // f_s_hat
  gemm(st, wsA, p[2], p[3], wsB, 65536, 256, 256);                      // qh
  gemm(st, wh,  p[0], p[1], kh, 320, 256, 256);                         // kh
  k_cu_caq<<<65536, 64, 0, st>>>(wsB, kh, wh, qm, mm, wsC);             // f_caq
  k_cu_fcq<<<65536, 256, 0, st>>>(wsA, wsC, sh, wsB, (size_t)65536 * 256); // f_cq
  k_cu_ac<<<16384, 128, 0, st>>>(wsB, wsA, mm, wsC);                    // f_cc_pre
  gemm(st, wsC, p[4], p[5], tgt, 65536, 512, 256);                      // c-proj
  k_cu_epi<<<131072, 256, 0, st>>>(tgt, fc, fm, fsu, mm, (size_t)65536 * 512);
}

// boundary unit: p = {kW,kb,qW,qb}
static void run_bu(hipStream_t st, const float* fb, const float* fm,
                   const float* fwu, const float* fsu,
                   const int* qm, const int* lm, const float* const* p,
                   float* qb, float* kw, float* baq, float* bq, float* outp) {
  gemm(st, fb,  p[2], p[3], qb, 512, 512, 512);
  gemm(st, fwu, p[0], p[1], kw, 320, 512, 512);
  dim3 ag(32, 8, 16);
  k_attn<<<ag, 256, 0, st>>>(qb, kw, fwu, baq, qm, nullptr, 32, 20, 0.125f);
  k_bu_fbq<<<1024, 256, 0, st>>>(fb, baq, fsu, lm, bq, (size_t)512 * 512);
  k_bu_out<<<512, 256, 0, st>>>(bq, fb, fm, fsu, lm, outp);
}

// moment unit: p = {fbW,fbb,fcW,fcb}
static void run_mu(hipStream_t st, const float* fcU, const float* fm, const float* fbU,
                   const int* mm, const float* const* p,
                   float* outer, float* meanb, float* conv, float* tgt) {
  k_outer<<<32768, 256, 0, st>>>(fbU, outer, (size_t)16384 * 512);
  gemm(st, outer, p[0], p[1], conv, 16384, 512, 512, 0);
  k_meank<<<32768, 256, 0, st>>>(fcU, meanb, (size_t)16384 * 512);
  gemm(st, meanb, p[2], p[3], conv, 16384, 512, 512, 1);
  k_mu_epi<<<32768, 256, 0, st>>>(conv, fm, mm, tgt, (size_t)16384 * 512);
}

// cga: p = {kW,kb,qW,qb,vW,vb}; out20 = query branch (LQ=20), outS = sentence
static void run_cga(hipStream_t st, const float* fw, const float* fs, const float* kv,
                    const int* qm, const int* mm, const float* const* p,
                    float* Kb, float* Vb, float* qp, float* qp2,
                    float* out20, float* outS) {
  gemm(st, kv, p[0], p[1], Kb, 16384, 512, 512);
  gemm(st, kv, p[4], p[5], Vb, 16384, 512, 512);
  gemm(st, fw, p[2], p[3], qp, 320, 512, 512);
  dim3 g1(20, 8, 16);
  k_attn<<<g1, 256, 0, st>>>(qp, Kb, Vb, out20, mm, qm, 20, 1024, 0.125f);
  gemm(st, fs, p[2], p[3], qp2, 16, 512, 512);
  dim3 g2(1, 8, 16);
  k_attn<<<g2, 256, 0, st>>>(qp2, Kb, Vb, outS, mm, nullptr, 1, 1024, 0.125f);
}

// ============================ entry point ==================================
extern "C" void kernel_launch(void* const* d_in, const int* in_sizes, int n_in,
                              void* d_out, int out_size, void* d_ws, size_t ws_size,
                              hipStream_t stream) {
  (void)in_sizes; (void)n_in; (void)out_size; (void)ws_size;
  const float* fc = (const float*)d_in[0];
  const float* fm = (const float*)d_in[1];
  const float* fb = (const float*)d_in[2];
  const float* fw = (const float*)d_in[3];
  const float* fs = (const float*)d_in[4];
  const int*   qm = (const int*)d_in[5];     // query_mask  (B*20)
  const int*   lm = (const int*)d_in[6];     // length_mask (B*32)
  const int*   mm = (const int*)d_in[7];     // moment_mask (B*1024)

  // params: jax tree-flatten = alphabetical DFS of nested dicts, (W,b) per lin.
  const float* P[52];
  for (int i = 0; i < 52; ++i) P[i] = (const float*)d_in[8 + i];
  const float* const* bu   = P + 0;   // bu : k.{W,b}, q.{W,b}
  const float* const* bu1  = P + 4;
  const float* const* cga  = P + 8;   // cga: k, q, v
  const float* const* cga1 = P + 14;
  const float* const* cu   = P + 20;  // cu : ak, aq, c, c_hat, s_hat, w_hat
  const float* const* cu1  = P + 32;
  const float* const* mu   = P + 44;  // mu : fb, fc
  const float* const* mu1  = P + 48;

  // workspace layout (floats); phases alias big buffers. Total ~376 MB.
  float* w    = (float*)d_ws;
  float* wsA  = w;                    // 16,777,216  f_c_hat | K-proj | conv acc
  float* wsB  = w + 16777216;         // 16,777,216  qh/f_cq | V-proj
  float* wsC  = w + 33554432;         // 16,777,216  f_caq/f_cc_pre | outer+meanK
  float* wsD  = w + 50331648;         // 33,554,432  fc2
  float* wsE  = w + 83886080;         //  8,388,608  fm2
  float* o_qp  = w + 92274688;        // 262,144
  float* o_qp2 = o_qp + 262144;       // 262,144
  float* o_wh  = o_qp2 + 262144;      //  81,920
  float* o_kh  = o_wh + 81920;        //  81,920
  float* o_sh  = o_kh + 81920;        //   4,096
  float* o_baq = o_sh + 4096;         // 262,144
  float* o_bq  = o_baq + 262144;      // 262,144
  float* o_fb2 = o_bq + 262144;       // 262,144
  float* o_fw2 = o_fb2 + 262144;      // 163,840
  float* o_fs2 = o_fw2 + 163840;      //   8,192

  float* out    = (float*)d_out;      // fc | fm | fb | fw | fs
  float* out_fc = out;
  float* out_fm = out_fc + 33554432;
  float* out_fb = out_fm + 8388608;
  float* out_fw = out_fb + 262144;
  float* out_fs = out_fw + 163840;

  // Phase A: fw1/fs1 = cga(fw/fs, fm)  -> straight into output slots
  run_cga(stream, fw, fs, fm, qm, mm, cga, wsA, wsB, o_qp, o_qp2, out_fw, out_fs);
  // Phase B: fc1 = content_unit(fc, fw1, fs1, fm)
  run_cu(stream, fc, fm, out_fw, out_fs, qm, mm, cu, wsA, wsB, wsC,
         o_wh, o_kh, o_sh, out_fc);
  // Phase C: fb1 = boundary_unit(fb, fw1, fs1, fm)
  run_bu(stream, fb, fm, out_fw, out_fs, qm, lm, bu, o_qp, o_qp2, o_baq, o_bq, out_fb);
  // Phase D: fm1 = moment_unit(fc1, fm, fb1)
  run_mu(stream, out_fc, fm, out_fb, mm, mu, wsC, wsC + 8388608, wsA, out_fm);
  // Phase E: fc2 = content_unit(fc, fw, fs, fm); out_fc = max(fc1, fc2)
  run_cu(stream, fc, fm, fw, fs, qm, mm, cu1, wsA, wsB, wsC,
         o_wh, o_kh, o_sh, wsD);
  k_max<<<131072, 256, 0, stream>>>(out_fc, wsD, (size_t)33554432);
  // Phase F: fb2 = boundary_unit(fb, fw, fs, fm); out_fb = max(fb1, fb2)
  run_bu(stream, fb, fm, fw, fs, qm, lm, bu1, o_qp, o_qp2, o_baq, o_bq, o_fb2);
  k_max<<<1024, 256, 0, stream>>>(out_fb, o_fb2, (size_t)262144);
  // Phase G: fm2 = moment_unit(fc2, fm, fb2); out_fm = max(fm1, fm2)
  run_mu(stream, wsD, fm, o_fb2, mm, mu1, wsC, wsC + 8388608, wsA, wsE);
  k_max<<<32768, 256, 0, stream>>>(out_fm, wsE, (size_t)8388608);
  // Phase H: fw2/fs2 = cga1(fw/fs, fm2); final maxes
  run_cga(stream, fw, fs, wsE, qm, mm, cga1, wsA, wsB, o_qp, o_qp2, o_fw2, o_fs2);
  k_max<<<640, 256, 0, stream>>>(out_fw, o_fw2, (size_t)163840);
  k_max<<<32, 256, 0, stream>>>(out_fs, o_fs2, (size_t)8192);
}
